// MultiHeadAttention_26104811225673
// MI455X (gfx1250) — matrix operand order repaired and checked by reading, not compiled, not run on hardware
//
#include <hip/hip_runtime.h>
#include <hip/hip_bf16.h>
#include <stdint.h>

// ---------------------------------------------------------------------------
// Multi-head attention for MI455X (gfx1250, wave32, WMMA).
//   B=4, N=2048, DIM=1024, H=16, Dh=64.
// Pass 0a: x (f32) -> bf16 row-major workspace.
// Pass 0b: Wq/Wk/Wv (f32) -> bf16 TRANSPOSED workspace Wt[n][k].
// Pass 1 : fused QKV projection GEMM (bf16 WMMA, f32 accum), 128x128 block
//          tiles, 8 waves; writes bf16 Q/K/Vt (Q pre-scaled, V transposed).
// Pass 2 : flash attention (online softmax), bf16 WMMA for QK^T and P*V.
// Workspace: 48MB QKV + 16MB x_bf16 + 6MB Wt  (~70MB total).
// ---------------------------------------------------------------------------

#define DIM_   1024
#define HEADS_ 16
#define DH_    64
#define SEQ_   2048
#define BATCH_ 4
#define ROWS_  (BATCH_ * SEQ_)                 // 8192
#define BH_    (BATCH_ * HEADS_)               // 64
#define QKV_ELEMS ((size_t)BH_ * SEQ_ * DH_)   // 8388608 per tensor
#define XB_ELEMS  ((size_t)ROWS_ * DIM_)       // 8388608
#define W_ELEMS   ((size_t)DIM_ * DIM_)        // 1048576

typedef __attribute__((ext_vector_type(16))) __bf16 v16bf;
typedef __attribute__((ext_vector_type(8)))  float  v8f;

union FragU {            // 32 bytes: two 16B loads -> one WMMA operand
    uint4 q[2];
    v16bf v;
};

// float -> bf16, round to nearest even (inputs are well-behaved normals)
static __device__ __forceinline__ unsigned short f2bf(float f) {
    uint32_t u = __float_as_uint(f);
    u += 0x7FFFu + ((u >> 16) & 1u);
    return (unsigned short)(u >> 16);
}

// ---- CDNA5 async global->LDS staging (GLOBAL_LOAD_ASYNC_TO_LDS_B128) ----
#if __has_builtin(__builtin_amdgcn_global_load_async_to_lds_b128)
#define HAVE_ASYNC_LDS 1
typedef int v4i_ __attribute__((vector_size(16)));
typedef v4i_ __attribute__((address_space(1))) as1_v4i;   // global
typedef v4i_ __attribute__((address_space(3))) as3_v4i;   // LDS
#endif

static __device__ __forceinline__ void stage_b128(const void* g, void* l) {
#ifdef HAVE_ASYNC_LDS
    __builtin_amdgcn_global_load_async_to_lds_b128(
        (as1_v4i*)g, (as3_v4i*)l, 0, 0);
#else
    *(uint4*)l = *(const uint4*)g;
#endif
}

static __device__ __forceinline__ void stage_wait() {
#ifdef HAVE_ASYNC_LDS
#if __has_builtin(__builtin_amdgcn_s_wait_asynccnt)
    __builtin_amdgcn_s_wait_asynccnt(0);
#else
    asm volatile("s_wait_asynccnt 0" ::: "memory");
#endif
#endif
}

// ---------------------------------------------------------------------------
// Pass 0a: convert x to bf16 (row-major). 8 elements / thread.
// ---------------------------------------------------------------------------
__global__ __launch_bounds__(256) void cvt_x_kernel(
    const float* __restrict__ x, unsigned short* __restrict__ xb)
{
    const size_t base = ((size_t)blockIdx.x * 256 + threadIdx.x) * 8;
    float4 a = *(const float4*)(x + base);
    float4 b = *(const float4*)(x + base + 4);
    unsigned short o[8] = { f2bf(a.x), f2bf(a.y), f2bf(a.z), f2bf(a.w),
                            f2bf(b.x), f2bf(b.y), f2bf(b.z), f2bf(b.w) };
    *(uint4*)(xb + base) = *(const uint4*)o;
}

// ---------------------------------------------------------------------------
// Pass 0b: convert + transpose W[k][n] (f32) -> Wt[n][k] (bf16), 64x64 tiles.
// ---------------------------------------------------------------------------
__global__ __launch_bounds__(256) void cvt_w_kernel(
    const float* __restrict__ Wq, const float* __restrict__ Wk,
    const float* __restrict__ Wv, unsigned short* __restrict__ wtq,
    unsigned short* __restrict__ wtk, unsigned short* __restrict__ wtv)
{
    __shared__ __align__(16) unsigned short T[64 * 72];
    const int z = blockIdx.z;
    const float* W = (z == 0) ? Wq : (z == 1) ? Wk : Wv;
    unsigned short* Wt = (z == 0) ? wtq : (z == 1) ? wtk : wtv;
    const int k0 = blockIdx.x * 64, n0 = blockIdx.y * 64;
    const int tid = threadIdx.x;

    for (int e = tid; e < 64 * 64; e += 256) {
        int r = e >> 6, c = e & 63;                       // r: k-row, c: n-col
        T[c * 72 + r] = f2bf(W[(size_t)(k0 + r) * DIM_ + n0 + c]);
    }
    __syncthreads();
    for (int e = tid; e < 512; e += 256) {                // 512 uint4 chunks
        int n = e >> 3, seg = e & 7;
        uint4 v = *(const uint4*)(T + n * 72 + seg * 8);
        *(uint4*)(Wt + (size_t)(n0 + n) * DIM_ + k0 + seg * 8) = v;
    }
}

// ---------------------------------------------------------------------------
// Pass 1: Y = x @ W + b. Block tile 128x128, K-step 64, 256 threads (8 waves:
// 2 in M x 4 in N). Each wave: 64x32 sub-tile = 4x2 WMMA accumulators,
// 16 v_wmma_f32_16x16x32_bf16 per K-iteration.
// ---------------------------------------------------------------------------
#define GS_STRIDE 72   // 128 x 64 bf16 tiles, padded (multiple of 8 -> 16B rows)

__global__ __launch_bounds__(256) void qkv_gemm_kernel(
    const unsigned short* __restrict__ xb,   // [8192][1024] bf16
    const unsigned short* __restrict__ wtq,  // [1024 n][1024 k] bf16
    const unsigned short* __restrict__ wtk,
    const unsigned short* __restrict__ wtv,
    const float* __restrict__ bq, const float* __restrict__ bk,
    const float* __restrict__ bv,
    unsigned short* __restrict__ q_ws,   // [BH][SEQ][DH] bf16, pre-scaled
    unsigned short* __restrict__ k_ws,   // [BH][SEQ][DH] bf16
    unsigned short* __restrict__ vt_ws)  // [BH][DH][SEQ] bf16 (transposed)
{
    __shared__ __align__(16) unsigned short As[128 * GS_STRIDE];
    __shared__ __align__(16) unsigned short Bs[128 * GS_STRIDE];

    const int tid    = threadIdx.x;
    const int wid    = tid >> 5;
    const int lane   = tid & 31;
    const int lane16 = lane & 15;
    const int grp    = lane >> 4;        // half-wave group
    const int waveM  = wid >> 2;         // 0..1 (64 rows each)
    const int waveN  = wid & 3;          // 0..3 (32 cols each)

    const int mode = blockIdx.z;         // 0=Q 1=K 2=V
    const unsigned short* Wt  = (mode == 0) ? wtq : (mode == 1) ? wtk : wtv;
    const float*          bias = (mode == 0) ? bq : (mode == 1) ? bk : bv;

    const int n0   = blockIdx.x * 128;   // output-feature tile
    const int row0 = blockIdx.y * 128;   // token-row tile

    v8f acc[4][2] = {};

    for (int k0 = 0; k0 < DIM_; k0 += 64) {
        __syncthreads();   // previous iteration's LDS reads complete
        // stage A (x rows) and B (Wt rows): 1024 uint4 each, 4 per thread
        for (int p4 = 0; p4 < 4; ++p4) {
            const int idx = tid + p4 * 256;
            const int row = idx >> 3, seg = idx & 7;
            stage_b128(xb + (size_t)(row0 + row) * DIM_ + k0 + seg * 8,
                       As + row * GS_STRIDE + seg * 8);
            stage_b128(Wt + (size_t)(n0 + row) * DIM_ + k0 + seg * 8,
                       Bs + row * GS_STRIDE + seg * 8);
        }
        stage_wait();
        __syncthreads();

        for (int c = 0; c < 2; ++c) {    // two 32-wide K chunks
            FragU a[4];
            for (int i = 0; i < 4; ++i) {
                const char* p = (const char*)As +
                    (size_t)((waveM * 64 + i * 16 + lane16) * GS_STRIDE +
                             c * 32 + grp * 8) * 2;
                a[i].q[0] = *(const uint4*)p;
                a[i].q[1] = *(const uint4*)(p + 32);   // K + 16
            }
            FragU b[2];
            for (int j = 0; j < 2; ++j) {
                const char* p = (const char*)Bs +
                    (size_t)((waveN * 32 + j * 16 + lane16) * GS_STRIDE +
                             c * 32 + grp * 8) * 2;
                b[j].q[0] = *(const uint4*)p;
                b[j].q[1] = *(const uint4*)(p + 32);
            }
            for (int i = 0; i < 4; ++i)
                for (int j = 0; j < 2; ++j)
                    acc[i][j] = __builtin_amdgcn_wmma_f32_16x16x32_bf16(
                        false, a[i].v, false, b[j].v, (short)0, acc[i][j],
                        false, false);
        }
    }

    // Epilogue: bias add, Q scaling, scatter into head-major bf16 layouts.
    const float qscale = 0.125f;  // Dh^-0.5, folded into Q
    for (int j = 0; j < 2; ++j) {
        const int col = n0 + waveN * 32 + j * 16 + lane16;   // 0..1023
        const int h = col >> 6, d = col & 63;
        const float bv_ = bias[col];
        for (int i = 0; i < 4; ++i) {
            for (int r = 0; r < 8; ++r) {
                const int row = row0 + waveM * 64 + i * 16 + r + grp * 8;
                const int bb = row >> 11, nq = row & (SEQ_ - 1);
                const int bh = bb * HEADS_ + h;
                float v = acc[i][j][r] + bv_;
                if (mode == 0) {
                    q_ws[((size_t)bh * SEQ_ + nq) * DH_ + d] = f2bf(v * qscale);
                } else if (mode == 1) {
                    k_ws[((size_t)bh * SEQ_ + nq) * DH_ + d] = f2bf(v);
                } else {
                    vt_ws[((size_t)bh * DH_ + d) * SEQ_ + nq] = f2bf(v);
                }
            }
        }
    }
}

// ---------------------------------------------------------------------------
// Pass 2: flash attention. grid = (SEQ/64 query tiles, BH). 4 waves; each
// wave owns 16 query rows. Loop over 64-key blocks with online softmax.
// ---------------------------------------------------------------------------
#define KS_STRIDE 72
#define VS_STRIDE 72
#define PS_STRIDE 72

__global__ __launch_bounds__(128) void flash_attn_kernel(
    const unsigned short* __restrict__ q_ws,
    const unsigned short* __restrict__ k_ws,
    const unsigned short* __restrict__ vt_ws,
    float* __restrict__ out)             // [B][SEQ][DIM] f32
{
    __shared__ __align__(16) unsigned short Ks[64 * KS_STRIDE];
    __shared__ __align__(16) unsigned short Vs[64 * VS_STRIDE];
    __shared__ __align__(16) unsigned short Ps[4][16 * PS_STRIDE];

    const int tid    = threadIdx.x;
    const int wid    = tid >> 5;
    const int lane   = tid & 31;
    const int lane16 = lane & 15;
    const int grp    = lane >> 4;

    const int bh = blockIdx.y;                  // 0..63
    const int q0 = blockIdx.x * 64 + wid * 16;  // this wave's first query row

    // Q fragments (A layout) straight from global, loaded once.
    FragU qf[2];
    {
        const unsigned short* qrow = q_ws + ((size_t)bh * SEQ_ + q0 + lane16) * DH_;
        for (int c = 0; c < 2; ++c) {
            const char* p = (const char*)(qrow + c * 32 + grp * 8);
            qf[c].q[0] = *(const uint4*)p;
            qf[c].q[1] = *(const uint4*)(p + 32);
        }
    }

    v8f   oacc[4] = {};
    float m_run[8], l_run[8];
    for (int r = 0; r < 8; ++r) { m_run[r] = -1e30f; l_run[r] = 0.0f; }

    for (int kb = 0; kb < SEQ_; kb += 64) {
        __syncthreads();   // previous block's LDS reads done
        // Stage K block [64 keys][64 dh] and Vt block [64 dh][64 keys]
        for (int p4 = 0; p4 < 4; ++p4) {
            const int idx = tid + p4 * 128;       // 512 uint4 per tile
            const int row = idx >> 3, seg = idx & 7;
            stage_b128(k_ws + ((size_t)bh * SEQ_ + kb + row) * DH_ + seg * 8,
                       Ks + row * KS_STRIDE + seg * 8);
            stage_b128(vt_ws + ((size_t)bh * DH_ + row) * SEQ_ + kb + seg * 8,
                       Vs + row * VS_STRIDE + seg * 8);
        }
        stage_wait();
        __syncthreads();

        // ---- S = Q * K^T : 4 key tiles x 2 dh chunks ----
        v8f sacc[4] = {};
        for (int j = 0; j < 4; ++j) {
            for (int c = 0; c < 2; ++c) {
                FragU kf;   // B layout: lane = key column, dh in table order (grp*8, +16)
                const char* p = (const char*)Ks +
                    (size_t)((j * 16 + lane16) * KS_STRIDE + c * 32 + grp * 8) * 2;
                kf.q[0] = *(const uint4*)p;
                kf.q[1] = *(const uint4*)(p + 32);
                sacc[j] = __builtin_amdgcn_wmma_f32_16x16x32_bf16(
                    false, qf[c].v, false, kf.v, (short)0, sacc[j], false, false);
            }
        }

        // ---- online softmax (row r lives in the 16 lanes sharing grp) ----
        float alpha[8];
        for (int r = 0; r < 8; ++r) {
            float mx = fmaxf(fmaxf(sacc[0][r], sacc[1][r]),
                             fmaxf(sacc[2][r], sacc[3][r]));
            mx = fmaxf(mx, __shfl_xor(mx, 1, 32));
            mx = fmaxf(mx, __shfl_xor(mx, 2, 32));
            mx = fmaxf(mx, __shfl_xor(mx, 4, 32));
            mx = fmaxf(mx, __shfl_xor(mx, 8, 32));
            const float m_new = fmaxf(m_run[r], mx);
            alpha[r] = __expf(m_run[r] - m_new);
            m_run[r] = m_new;
            float ls = 0.0f;
            for (int j = 0; j < 4; ++j) {
                float pv = __expf(sacc[j][r] - m_new);
                sacc[j][r] = pv;
                ls += pv;
            }
            ls += __shfl_xor(ls, 1, 32);
            ls += __shfl_xor(ls, 2, 32);
            ls += __shfl_xor(ls, 4, 32);
            ls += __shfl_xor(ls, 8, 32);
            l_run[r] = l_run[r] * alpha[r] + ls;
        }
        for (int t = 0; t < 4; ++t)
            for (int r = 0; r < 8; ++r)
                oacc[t][r] *= alpha[r];

        // ---- transpose P (C layout) -> A layout via per-wave LDS tile ----
        for (int j = 0; j < 4; ++j)
            for (int r = 0; r < 8; ++r)
                Ps[wid][(r + grp * 8) * PS_STRIDE + j * 16 + lane16] = f2bf(sacc[j][r]);
        __syncthreads();

        // ---- O += P * V : 4 dh tiles x 2 key chunks ----
        FragU pf[2];
        for (int c = 0; c < 2; ++c) {
            const char* p = (const char*)Ps[wid] +
                (size_t)(lane16 * PS_STRIDE + c * 32 + grp * 8) * 2;
            pf[c].q[0] = *(const uint4*)p;
            pf[c].q[1] = *(const uint4*)(p + 32);
        }
        for (int t = 0; t < 4; ++t) {
            for (int c = 0; c < 2; ++c) {
                FragU vf;   // B layout: lane = dh column, keys in table order (grp*8, +16)
                const char* p = (const char*)Vs +
                    (size_t)((t * 16 + lane16) * VS_STRIDE + c * 32 + grp * 8) * 2;
                vf.q[0] = *(const uint4*)p;
                vf.q[1] = *(const uint4*)(p + 32);
                oacc[t] = __builtin_amdgcn_wmma_f32_16x16x32_bf16(
                    false, pf[c].v, false, vf.v, (short)0, oacc[t], false, false);
            }
        }
    }

    // ---- finalize: divide by running sum, write f32 output [b][nq][h*64+d] ----
    const int bb = bh >> 4, h = bh & 15;
    for (int t = 0; t < 4; ++t) {
        const int d = t * 16 + lane16;
        for (int r = 0; r < 8; ++r) {
            const int nq = q0 + r + grp * 8;
            out[((size_t)bb * SEQ_ + nq) * DIM_ + h * DH_ + d] = oacc[t][r] / l_run[r];
        }
    }
}

// ---------------------------------------------------------------------------
extern "C" void kernel_launch(void* const* d_in, const int* in_sizes, int n_in,
                              void* d_out, int out_size, void* d_ws, size_t ws_size,
                              hipStream_t stream) {
    const float* x  = (const float*)d_in[0];
    const float* Wq = (const float*)d_in[1];
    const float* bq = (const float*)d_in[2];
    const float* Wk = (const float*)d_in[3];
    const float* bk = (const float*)d_in[4];
    const float* Wv = (const float*)d_in[5];
    const float* bv = (const float*)d_in[6];

    unsigned short* q_ws  = (unsigned short*)d_ws;
    unsigned short* k_ws  = q_ws + QKV_ELEMS;
    unsigned short* vt_ws = k_ws + QKV_ELEMS;
    unsigned short* xb    = vt_ws + QKV_ELEMS;
    unsigned short* wtq   = xb + XB_ELEMS;
    unsigned short* wtk   = wtq + W_ELEMS;
    unsigned short* wtv   = wtk + W_ELEMS;

    // Pass 0: one-time bf16 conversion (x) and conversion+transpose (W).
    cvt_x_kernel<<<XB_ELEMS / (256 * 8), 256, 0, stream>>>(x, xb);
    cvt_w_kernel<<<dim3(DIM_ / 64, DIM_ / 64, 3), 256, 0, stream>>>(
        Wq, Wk, Wv, wtq, wtk, wtv);

    // Pass 1: QKV projection, 128x128 tiles: (8 n-tiles, 64 row-tiles, 3)
    qkv_gemm_kernel<<<dim3(DIM_ / 128, ROWS_ / 128, 3), 256, 0, stream>>>(
        xb, wtq, wtk, wtv, bq, bk, bv, q_ws, k_ws, vt_ws);

    // Pass 2: flash attention: 32 query tiles x 64 (batch*head)
    flash_attn_kernel<<<dim3(SEQ_ / 64, BH_), 128, 0, stream>>>(
        q_ws, k_ws, vt_ws, (float*)d_out);
}
